// DecoderModel_59974923322051
// MI455X (gfx1250) — compile-verified, matrix operand order checked by reading, not run on hardware
//
#include <hip/hip_runtime.h>
#include <hip/hip_bf16.h>

#define HID     128
#define EMB     128
#define LAYERS  4
#define VOCAB   32000
#define SRC_LEN 64
#define BATCH   128
#define TSTEPS  32
#define GATES   512   // 4*HID
#define KCAT    256   // EMB + HID

// phase-2 tiling
#define MSUB    2                 // 16-row M subtiles per block -> M=32
#define MROWS   (16 * MSUB)
#define P2_THREADS 512
#define P2_WAVES   (P2_THREADS / 32)

typedef __attribute__((ext_vector_type(16))) __bf16 v16bf;
typedef __attribute__((ext_vector_type(8)))  __bf16 v8bf;
typedef __attribute__((ext_vector_type(8)))  float  v8f;

// Load one 16x32 bf16 WMMA fragment from a row-major [16][ld] block.
// A-matrix: lane&15 = row M, lane>>4 selects K-halves {hi*8.., 16+hi*8..}.
// B-matrix (32x16, K x N) mirrors with lane&15 = col N — matches row-major
// w_out[n][k] (K contiguous) exactly. Lowers to two global/ds b128 loads.
__device__ __forceinline__ v16bf load_frag(const __bf16* base, int ld, int lane) {
  const int row = lane & 15;
  const int hi  = lane >> 4;
  const __bf16* p = base + row * ld + hi * 8;
  v8bf c0 = *(const v8bf*)(p);
  v8bf c1 = *(const v8bf*)(p + 16);
  v16bf f;
#pragma unroll
  for (int i = 0; i < 8; ++i) { f[i] = c0[i]; f[i + 8] = c1[i]; }
  return f;
}

__device__ __forceinline__ float sigmf(float x) { return 1.f / (1.f + __expf(-x)); }

// ---------------- prep kernels: build bf16 weights in workspace ----------------
__global__ void conv_wout_k(const float* __restrict__ w, __bf16* __restrict__ o, int n) {
  int i = blockIdx.x * 256 + threadIdx.x;
  if (i < n) o[i] = (__bf16)w[i];
}

__global__ void build_wcat_k(const float* __restrict__ wi, const float* __restrict__ wh,
                             __bf16* __restrict__ o) {
  int i = blockIdx.x * 256 + threadIdx.x;
  if (i >= LAYERS * GATES * KCAT) return;
  int l = i / (GATES * KCAT);
  int r = (i / KCAT) % GATES;
  int k = i % KCAT;
  float v = (k < EMB) ? wi[(l * GATES + r) * EMB + k]
                      : wh[(l * GATES + r) * HID + (k - EMB)];
  o[i] = (__bf16)v;
}

__global__ void build_bsum_k(const float* __restrict__ bi, const float* __restrict__ bh,
                             float* __restrict__ o) {
  int i = blockIdx.x * 256 + threadIdx.x;
  if (i < LAYERS * GATES) o[i] = bi[i] + bh[i];
}

// ---------------- phase 1: LSTM recurrence + attention (parallel over batch) ----
// 8 blocks x 16 batch rows; 8 waves/block. Gate GEMM: [16 x 512] = Acat[16x256] @ Wcat^T
__global__ __launch_bounds__(256)
void lstm_phase1(const float* __restrict__ enc_h,
                 const float* __restrict__ hidden0,
                 const float* __restrict__ cell0,
                 const float* __restrict__ embedding,
                 const int*   __restrict__ start_code,
                 const __bf16* __restrict__ wcat,
                 const float* __restrict__ bsum,
                 __bf16* __restrict__ hc)
{
  __shared__ float  sh_h[LAYERS][16][HID];
  __shared__ float  sh_c[LAYERS][16][HID];
  __shared__ float  sh_g[16][GATES];
  __shared__ __bf16 sh_a[16][KCAT];
  __shared__ float  sh_e[SRC_LEN][16];
  __shared__ float  sh_x[EMB];

  const int tid  = threadIdx.x;
  const int lane = tid & 31;
  const int wave = tid >> 5;
  const int b0   = blockIdx.x * 16;

  for (int idx = tid; idx < LAYERS * 16 * HID; idx += 256) {
    int l = idx / (16 * HID);
    int r = (idx / HID) & 15;
    int k = idx & (HID - 1);
    sh_h[l][r][k] = hidden0[((size_t)l * BATCH + b0 + r) * HID + k];
    sh_c[l][r][k] = cell0 [((size_t)l * BATCH + b0 + r) * HID + k];
  }
  if (tid < EMB) {
    int tok = start_code[0];
    sh_x[tid] = embedding[(size_t)tok * EMB + tid];
  }
  __syncthreads();

  for (int t = 0; t < TSTEPS; ++t) {
    for (int l = 0; l < LAYERS; ++l) {
      // Acat = [x_in | h_l] as bf16
      for (int idx = tid; idx < 16 * KCAT; idx += 256) {
        int r = idx >> 8;
        int k = idx & 255;
        float v = (k < HID) ? ((l == 0) ? sh_x[k] : sh_h[l - 1][r][k])
                            : sh_h[l][r][k - HID];
        sh_a[r][k] = (__bf16)v;
      }
      __syncthreads();

      // gates[16][512] via WMMA, wave w owns n in [w*64, w*64+64)
      const __bf16* wl = wcat + (size_t)l * GATES * KCAT;
#pragma unroll
      for (int j = 0; j < 4; ++j) {
        const int nbase = (wave * 4 + j) * 16;
        v8f acc = {};
#pragma unroll
        for (int kk = 0; kk < 8; ++kk) {
          v16bf a = load_frag(&sh_a[0][kk * 32], KCAT, lane);
          v16bf b = load_frag(wl + (size_t)nbase * KCAT + kk * 32, KCAT, lane);
          acc = __builtin_amdgcn_wmma_f32_16x16x32_bf16(false, a, false, b,
                                                        (short)0, acc, false, false);
        }
        const int n  = nbase + (lane & 15);
        const int hi = lane >> 4;
#pragma unroll
        for (int r = 0; r < 8; ++r) sh_g[r + 8 * hi][n] = acc[r];
      }
      __syncthreads();

      // elementwise LSTM cell update (gate order i,f,g,o)
      const float* bl = bsum + l * GATES;
      for (int idx = tid; idx < 16 * HID; idx += 256) {
        int r = idx >> 7;
        int u = idx & (HID - 1);
        float gi = sh_g[r][u]           + bl[u];
        float gf = sh_g[r][HID + u]     + bl[HID + u];
        float gg = sh_g[r][2 * HID + u] + bl[2 * HID + u];
        float go = sh_g[r][3 * HID + u] + bl[3 * HID + u];
        float cn = sigmf(gf) * sh_c[l][r][u] + sigmf(gi) * tanhf(gg);
        sh_c[l][r][u] = cn;
        sh_h[l][r][u] = sigmf(go) * tanhf(cn);
      }
      __syncthreads();
    }

    // attention scores e[s][b] = h_top[b] . enc_h[s][b0+b]
    for (int p = tid; p < SRC_LEN * 16; p += 256) {
      int s = p >> 4;
      int b = p & 15;
      const float* eh = enc_h + ((size_t)s * BATCH + b0 + b) * HID;
      const float* ht = sh_h[LAYERS - 1][b];
      float d = 0.f;
#pragma unroll 4
      for (int k = 0; k < HID; k += 4) {
        float4 v = *(const float4*)(eh + k);
        d += v.x * ht[k] + v.y * ht[k + 1] + v.z * ht[k + 2] + v.w * ht[k + 3];
      }
      sh_e[s][b] = d;
    }
    __syncthreads();

    // softmax over source positions (per batch column)
    if (tid < 16) {
      float m = -1e30f;
      for (int s = 0; s < SRC_LEN; ++s) m = fmaxf(m, sh_e[s][tid]);
      float sum = 0.f;
      for (int s = 0; s < SRC_LEN; ++s) {
        float v = __expf(sh_e[s][tid] - m);
        sh_e[s][tid] = v;
        sum += v;
      }
      float inv = 1.f / sum;
      for (int s = 0; s < SRC_LEN; ++s) sh_e[s][tid] *= inv;
    }
    __syncthreads();

    // h_c = [h_top | ctx] -> workspace (bf16, A-matrix for phase 2)
    for (int idx = tid; idx < 16 * KCAT; idx += 256) {
      int r = idx >> 8;
      int k = idx & 255;
      float v;
      if (k < HID) {
        v = sh_h[LAYERS - 1][r][k];
      } else {
        int h = k - HID;
        float acc = 0.f;
        for (int s = 0; s < SRC_LEN; ++s)
          acc += sh_e[s][r] * enc_h[((size_t)s * BATCH + b0 + r) * HID + h];
        v = acc;
      }
      hc[((size_t)(t * BATCH + b0 + r)) * KCAT + k] = (__bf16)v;
    }
    __syncthreads();
  }
}

// ---------------- phase 2: [4096 x 32000] bf16 WMMA GEMM + ReLU + softmax -------
// Block owns MROWS=32 full rows (2 M-subtiles share every B fragment -> 2x L2
// reuse on w_out). Softmax sums stay block-local (LDS f32 atomics). 16 waves.
__global__ __launch_bounds__(P2_THREADS)
void gemm_softmax(const __bf16* __restrict__ hc,
                  const __bf16* __restrict__ wout,
                  const float*  __restrict__ b_out,
                  float* __restrict__ out)
{
  __shared__ __bf16 sh_a[MROWS][KCAT];     // 16 KB
  __shared__ float  sh_sum[MROWS];

  const int tid  = threadIdx.x;
  const int lane = tid & 31;
  const int wave = tid >> 5;
  const int row0 = blockIdx.x * MROWS;     // over TSTEPS*BATCH = 4096 rows

  // stage A tile [32][256] bf16 into LDS
  for (int idx = tid; idx < (MROWS * KCAT) / 8; idx += P2_THREADS)
    ((v8bf*)&sh_a[0][0])[idx] = ((const v8bf*)(hc + (size_t)row0 * KCAT))[idx];
  if (tid < MROWS) sh_sum[tid] = 0.f;
  __syncthreads();

  const int hi = lane >> 4;
  const int nl = lane & 15;
  float sloc[MSUB][8];
#pragma unroll
  for (int m = 0; m < MSUB; ++m)
#pragma unroll
    for (int r = 0; r < 8; ++r) sloc[m][r] = 0.f;

  for (int tile = wave; tile < VOCAB / 16; tile += P2_WAVES) {  // 125 per wave
    const int nbase = tile * 16;
    v8f acc[MSUB];
#pragma unroll
    for (int m = 0; m < MSUB; ++m) acc[m] = (v8f){};

#pragma unroll
    for (int kk = 0; kk < 8; ++kk) {
      // one B fragment feeds MSUB WMMAs (2x reuse of L2-resident w_out)
      v16bf b = load_frag(wout + (size_t)nbase * KCAT + kk * 32, KCAT, lane);
#pragma unroll
      for (int m = 0; m < MSUB; ++m) {
        v16bf a = load_frag(&sh_a[m * 16][kk * 32], KCAT, lane);
        acc[m] = __builtin_amdgcn_wmma_f32_16x16x32_bf16(false, a, false, b,
                                                         (short)0, acc[m], false, false);
      }
    }

    const int   n    = nbase + nl;
    const float bias = b_out[n];
#pragma unroll
    for (int m = 0; m < MSUB; ++m) {
#pragma unroll
      for (int r = 0; r < 8; ++r) {
        float z = fmaxf(acc[m][r] + bias, 0.f);   // Linear + ReLU
        float e = __expf(z);                      // ReLU'd logits are small: safe
        out[(size_t)(row0 + m * 16 + r + 8 * hi) * VOCAB + n] = e;
        sloc[m][r] += e;
      }
    }
  }
#pragma unroll
  for (int m = 0; m < MSUB; ++m)
#pragma unroll
    for (int r = 0; r < 8; ++r)
      atomicAdd(&sh_sum[m * 16 + r + 8 * hi], sloc[m][r]);
  __syncthreads();

  // normalize in place (float4; re-read is L2-hot: 4 MB/block just written)
  for (int idx = tid; idx < (MROWS * VOCAB) / 4; idx += P2_THREADS) {
    int   r = idx / (VOCAB / 4);
    float s = 1.f / sh_sum[r];
    float4* p = (float4*)(out + (size_t)row0 * VOCAB) + idx;
    float4 v = *p;
    v.x *= s; v.y *= s; v.z *= s; v.w *= s;
    *p = v;
  }
}

// ------------------------------- host launcher ---------------------------------
extern "C" void kernel_launch(void* const* d_in, const int* in_sizes, int n_in,
                              void* d_out, int out_size, void* d_ws, size_t ws_size,
                              hipStream_t stream)
{
  const float* enc_h  = (const float*)d_in[0];
  const float* hidden = (const float*)d_in[1];
  const float* cell   = (const float*)d_in[2];
  const float* emb    = (const float*)d_in[3];
  const float* w_ih   = (const float*)d_in[4];
  const float* w_hh   = (const float*)d_in[5];
  const float* b_ih   = (const float*)d_in[6];
  const float* b_hh   = (const float*)d_in[7];
  const float* w_out  = (const float*)d_in[8];
  const float* b_out  = (const float*)d_in[9];
  const int*   start  = (const int*)d_in[11];
  float* out = (float*)d_out;

  // workspace layout (~19.6 MB): bf16 w_out | bf16 [Wi|Wh] | f32 bias sums | bf16 h_c
  char* ws = (char*)d_ws;
  size_t o = 0;
  __bf16* wout_bf = (__bf16*)(ws + o);  o += (size_t)VOCAB * KCAT * 2;          o = (o + 255) & ~(size_t)255;
  __bf16* wcat_bf = (__bf16*)(ws + o);  o += (size_t)LAYERS * GATES * KCAT * 2; o = (o + 255) & ~(size_t)255;
  float*  bsum    = (float*) (ws + o);  o += (size_t)LAYERS * GATES * 4;        o = (o + 255) & ~(size_t)255;
  __bf16* hc      = (__bf16*)(ws + o);

  conv_wout_k <<<(VOCAB * KCAT + 255) / 256, 256, 0, stream>>>(w_out, wout_bf, VOCAB * KCAT);
  build_wcat_k<<<(LAYERS * GATES * KCAT + 255) / 256, 256, 0, stream>>>(w_ih, w_hh, wcat_bf);
  build_bsum_k<<<(LAYERS * GATES + 255) / 256, 256, 0, stream>>>(b_ih, b_hh, bsum);

  lstm_phase1 <<<BATCH / 16, 256, 0, stream>>>(enc_h, hidden, cell, emb, start,
                                               wcat_bf, bsum, hc);
  gemm_softmax<<<(TSTEPS * BATCH) / MROWS, P2_THREADS, 0, stream>>>(hc, wout_bf, b_out, out);
}
